// LinearAttentionLayer_25786983646112
// MI455X (gfx1250) — matrix operand order repaired and checked by reading, not compiled, not run on hardware
//
#include <hip/hip_runtime.h>

#define DIM  1024
#define TSEQ 4096
#define NB   8

typedef __attribute__((ext_vector_type(16))) _Float16 v16h;
typedef __attribute__((ext_vector_type(8)))  float    v8f;
typedef __attribute__((ext_vector_type(4)))  int      v4i;

union FragH { v16h h; uint4 q[2]; };

__device__ __forceinline__ float elu1f(float x) { return x > 0.f ? x + 1.f : __expf(x); }

// one v_cvt_pk_rtz_f16_f32: packs two f32 into two f16
__device__ __forceinline__ unsigned int pkrtz(float a, float b) {
    auto h2 = __builtin_amdgcn_cvt_pkrtz(a, b);
    return __builtin_bit_cast(unsigned int, h2);
}

// ---- CDNA5 async global->LDS copy (no VGPR staging, ASYNCcnt tracked) -----
#if defined(__has_builtin)
#  if __has_builtin(__builtin_amdgcn_global_load_async_to_lds_b128)
#    define ASYNC_LDS 1
#  endif
#endif
#ifndef ASYNC_LDS
#  define ASYNC_LDS 0
#endif

__device__ __forceinline__ void cp16_g2l(unsigned short* l, const _Float16* g) {
#if ASYNC_LDS
    __builtin_amdgcn_global_load_async_to_lds_b128(
        (__attribute__((address_space(1))) v4i*)(g),
        (__attribute__((address_space(3))) v4i*)(unsigned int)(size_t)l,
        0, 0);
#else
    *(uint4*)l = *(const uint4*)g;
#endif
}
__device__ __forceinline__ void wait_async() {
#if ASYNC_LDS
    asm volatile("s_wait_asynccnt 0x0" ::: "memory");
#endif
}

// ---------------------------------------------------------------------------
// Tiled WMMA GEMM:  C = epilogue( A @ B^T )
//   A: [M,K] row-major f16, per-batch stride
//   B: [N,K] row-major f16 ("B-transposed" storage)
//   Macrotile: 128(M) x 128(N), K-step 32; 256 threads = 8 waves (4M x 2N),
//   each wave computes 32(M) x 64(N) = 8 WMMA accumulators.
//   Double-buffered LDS; async global->LDS copies overlap the WMMA phase.
// MODE 0: Q proj   : elu1(x+bias), f16 row-major out (LDS restage)
// MODE 1: K proj   : elu1(x+bias), f16 TRANSPOSED out C^T[n][m]
// MODE 2: V proj   : x+bias,       f16 TRANSPOSED out
// MODE 3: KV       : x,            f16 TRANSPOSED out
// MODE 4: attn_num : x/(Q*Ksum+eps), f16 row-major out (LDS restage)
// MODE 5: out proj : x+bias+residual, f32 row-major out
// ---------------------------------------------------------------------------
template <int MODE>
__global__ __launch_bounds__(256) void gemm_wmma(
    const _Float16* __restrict__ Ap, long long aStride, int lda,
    const _Float16* __restrict__ Bt, long long bStride, int ldb,
    void* __restrict__ Cv, long long cStride, int ldc,
    const float* __restrict__ bias,
    const float* __restrict__ extra, long long extraStride,
    int kIters)
{
    extern __shared__ char smem[];
    // buf b (b=0,1): A tile at smem + b*20480, B tile at +10240 (bytes)
    unsigned short* s_base = (unsigned short*)smem;
    unsigned short* s_e    = (unsigned short*)smem;   // epilogue restage (reuse)

    const int z    = blockIdx.z;
    const int m0   = blockIdx.y * 128;
    const int n0   = blockIdx.x * 128;
    const int tid  = threadIdx.x;
    const int lane = tid & 31;
    const int wave = tid >> 5;
    const int mw   = wave & 3;    // 4 waves along M
    const int nw   = wave >> 2;   // 2 waves along N
    const int mbase = m0 + 32 * mw;
    const int nbase = n0 + 64 * nw;

    const _Float16* Ah = Ap + (size_t)((long long)z * aStride);
    const _Float16* Bp = Bt + (size_t)((long long)z * bStride);

    v8f acc[2][4];
#pragma unroll
    for (int mi = 0; mi < 2; mi++)
#pragma unroll
        for (int c = 0; c < 4; c++)
#pragma unroll
            for (int i = 0; i < 8; i++) acc[mi][c][i] = 0.f;

    const int r4 = tid >> 2, sg4 = (tid & 3) * 8;
    // per-thread global sources (rows r4 and r4+64 of the A / B tiles)
    const _Float16* gA0 = Ah + (size_t)(m0 + r4) * lda + sg4;
    const _Float16* gA1 = Ah + (size_t)(m0 + r4 + 64) * lda + sg4;
    const _Float16* gB0 = Bp + (size_t)(n0 + r4) * ldb + sg4;
    const _Float16* gB1 = Bp + (size_t)(n0 + r4 + 64) * ldb + sg4;
    // per-thread LDS destinations (within buffer 0, shorts)
    unsigned short* lA0 = s_base + r4 * 40 + sg4;
    unsigned short* lA1 = s_base + (r4 + 64) * 40 + sg4;
    unsigned short* lB0 = s_base + 5120 + r4 * 40 + sg4;          // +10240B
    unsigned short* lB1 = s_base + 5120 + (r4 + 64) * 40 + sg4;

    auto issue_tile = [&](int kk) {       // stage tile kk into buffer kk&1
        const int k0   = kk * 32;
        const int boff = (kk & 1) * 10240;  // shorts: 20480 bytes per buffer
        cp16_g2l(lA0 + boff, gA0 + k0);
        cp16_g2l(lA1 + boff, gA1 + k0);
        cp16_g2l(lB0 + boff, gB0 + k0);
        cp16_g2l(lB1 + boff, gB1 + k0);
    };

    // fragment base pointers within buffer 0 (shorts)
    const unsigned short* pa0 = s_base + (32 * mw + (lane & 15)) * 40 + ((lane >> 4) * 16);
    const unsigned short* pb0 = s_base + 5120 + (64 * nw + (lane & 15)) * 40 + ((lane >> 4) * 16);

    issue_tile(0);
    for (int kk = 0; kk < kIters; ++kk) {
        wait_async();             // tile kk landed (this wave's copies)
        __syncthreads();          // all waves' copies visible; other buffer free
        if (kk + 1 < kIters) issue_tile(kk + 1);   // overlaps with compute below

        const int boff = (kk & 1) * 10240;
        const unsigned short* pa = pa0 + boff;
        const unsigned short* pb = pb0 + boff;

        // ---- A fragments: lanes 0-15 K={0..15}, lanes 16-31 K={16..31}
        FragH fa[2], fb[2];
        fa[0].q[0] = *(const uint4*)(pa);
        fa[0].q[1] = *(const uint4*)(pa + 8);
        fa[1].q[0] = *(const uint4*)(pa + 16 * 40);
        fa[1].q[1] = *(const uint4*)(pa + 16 * 40 + 8);
        fb[0].q[0] = *(const uint4*)(pb);
        fb[0].q[1] = *(const uint4*)(pb + 8);
#pragma unroll
        for (int c = 0; c < 4; c++) {
            if (c < 3) {  // prefetch next B fragment while WMMAs run
                const unsigned short* pbn = pb + (16 * (c + 1)) * 40;
                fb[(c + 1) & 1].q[0] = *(const uint4*)(pbn);
                fb[(c + 1) & 1].q[1] = *(const uint4*)(pbn + 8);
            }
            acc[0][c] = __builtin_amdgcn_wmma_f32_16x16x32_f16(
                false, fa[0].h, false, fb[c & 1].h, (short)0, acc[0][c], false, false);
            acc[1][c] = __builtin_amdgcn_wmma_f32_16x16x32_f16(
                false, fa[1].h, false, fb[c & 1].h, (short)0, acc[1][c], false, false);
        }
    }

    __syncthreads();

    float bval[4];
    if (MODE == 0 || MODE == 1 || MODE == 2 || MODE == 5) {
#pragma unroll
        for (int c = 0; c < 4; c++) bval[c] = bias[nbase + 16 * c + (lane & 15)];
    }

    // C fragment: lane (l<16): N=l, M=8*(l>=16)+r for VGPR r
    if constexpr (MODE == 0 || MODE == 4) {
        const float* Ks = (MODE == 4) ? (extra + (size_t)((long long)z * extraStride)) : nullptr;
        unsigned short* se = s_e + wave * 32 * 72;
#pragma unroll
        for (int c = 0; c < 4; c++) {
            const int n = nbase + 16 * c + (lane & 15);
            const float ks = (MODE == 4) ? Ks[n] : 0.f;
#pragma unroll
            for (int mi = 0; mi < 2; mi++) {
#pragma unroll
                for (int r = 0; r < 8; r++) {
                    float v = acc[mi][c][r];
                    if (MODE == 0) {
                        v = elu1f(v + bval[c]);
                    } else {
                        const int m = mbase + 16 * mi + 8 * (lane >> 4) + r;
                        const float q = (float)Ah[(size_t)m * lda + n];
                        v = v / (q * ks + 1e-6f);
                    }
                    _Float16 h = (_Float16)v;
                    se[(16 * mi + 8 * (lane >> 4) + r) * 72 + 16 * c + (lane & 15)] =
                        __builtin_bit_cast(unsigned short, h);
                }
            }
        }
        __syncthreads();
        _Float16* Ch = (_Float16*)Cv + (size_t)((long long)z * cStride);
#pragma unroll
        for (int i = 0; i < 8; i++) {
            const int ch = i * 32 + lane;
            const int rr = ch >> 3, sg = (ch & 7) * 8;
            uint4 v = *(const uint4*)(se + rr * 72 + sg);
            *(uint4*)(Ch + (size_t)(mbase + rr) * ldc + nbase + sg) = v;
        }
    } else if constexpr (MODE == 5) {
        float* Cf = (float*)Cv + (size_t)((long long)z * cStride);
        const float* res = extra + (size_t)((long long)z * extraStride);
#pragma unroll
        for (int c = 0; c < 4; c++) {
            const int n = nbase + 16 * c + (lane & 15);
#pragma unroll
            for (int mi = 0; mi < 2; mi++) {
#pragma unroll
                for (int r = 0; r < 8; r++) {
                    const int m = mbase + 16 * mi + 8 * (lane >> 4) + r;
                    Cf[(size_t)m * ldc + n] = acc[mi][c][r] + bval[c] + res[(size_t)m * ldc + n];
                }
            }
        }
    } else {
        // MODE 1,2,3: transposed f16 store C^T[n][m] — 8 M-consecutive values per lane
        _Float16* Ct = (_Float16*)Cv + (size_t)((long long)z * cStride);
#pragma unroll
        for (int c = 0; c < 4; c++) {
            const int n = nbase + 16 * c + (lane & 15);
#pragma unroll
            for (int mi = 0; mi < 2; mi++) {
                const int mseg = mbase + 16 * mi + 8 * (lane >> 4);
                float v[8];
#pragma unroll
                for (int r = 0; r < 8; r++) {
                    float x = acc[mi][c][r];
                    if (MODE == 1) x = elu1f(x + bval[c]);
                    if (MODE == 2) x = x + bval[c];
                    v[r] = x;
                }
                uint4 u = make_uint4(pkrtz(v[0], v[1]), pkrtz(v[2], v[3]),
                                     pkrtz(v[4], v[5]), pkrtz(v[6], v[7]));
                *(uint4*)(Ct + (size_t)n * ldc + mseg) = u;
            }
        }
    }
}

// ---- f32 -> f16 conversion (weights / activations) ------------------------
__global__ void cvt_f16(const float* __restrict__ s, _Float16* __restrict__ d, int n4)
{
    const int i = blockIdx.x * blockDim.x + threadIdx.x;
    if (i < n4) {
        float4 v = ((const float4*)s)[i];
        uint2 u;
        u.x = pkrtz(v.x, v.y);
        u.y = pkrtz(v.z, v.w);
        ((uint2*)d)[i] = u;
    }
}

// ---- K_sum[b][d] = sum_t K^T[b][d][t] -------------------------------------
__global__ __launch_bounds__(256) void ksum_kernel(const _Float16* __restrict__ Kt,
                                                   float* __restrict__ Ks)
{
    const int row  = blockIdx.x * 8 + (threadIdx.x >> 5);
    const int lane = threadIdx.x & 31;
    const uint4* r4 = (const uint4*)(Kt + (size_t)row * TSEQ);
    float s = 0.f;
    for (int i = 0; i < 16; i++) {
        union { uint4 u; _Float16 h[8]; } t;
        t.u = r4[lane + 32 * i];
#pragma unroll
        for (int j = 0; j < 8; j++) s += (float)t.h[j];
    }
#pragma unroll
    for (int off = 16; off; off >>= 1) s += __shfl_down(s, off);
    if (lane == 0) Ks[row] = s;
}

// ---- in-place LayerNorm over last dim (wave per row, wave32) --------------
__global__ __launch_bounds__(256) void ln_kernel(float* __restrict__ out,
                                                 const float* __restrict__ gamma,
                                                 const float* __restrict__ beta)
{
    const int row  = blockIdx.x * 8 + (threadIdx.x >> 5);
    const int lane = threadIdx.x & 31;
    float4* p = (float4*)(out + (size_t)row * DIM);
    float4 v[8];
    float s = 0.f, s2 = 0.f;
#pragma unroll
    for (int i = 0; i < 8; i++) {
        v[i] = p[lane + 32 * i];
        s  += v[i].x + v[i].y + v[i].z + v[i].w;
        s2 += v[i].x * v[i].x + v[i].y * v[i].y + v[i].z * v[i].z + v[i].w * v[i].w;
    }
#pragma unroll
    for (int off = 16; off; off >>= 1) {
        s  += __shfl_down(s, off);
        s2 += __shfl_down(s2, off);
    }
    s  = __shfl(s, 0);
    s2 = __shfl(s2, 0);
    const float mu  = s * (1.f / DIM);
    const float var = s2 * (1.f / DIM) - mu * mu;
    const float rs  = rsqrtf(var + 1e-5f);
    const float4* g4 = (const float4*)gamma;
    const float4* b4 = (const float4*)beta;
#pragma unroll
    for (int i = 0; i < 8; i++) {
        float4 g = g4[lane + 32 * i], b = b4[lane + 32 * i], w = v[i];
        w.x = (w.x - mu) * rs * g.x + b.x;
        w.y = (w.y - mu) * rs * g.y + b.y;
        w.z = (w.z - mu) * rs * g.z + b.z;
        w.w = (w.w - mu) * rs * g.w + b.w;
        p[lane + 32 * i] = w;
    }
}

extern "C" void kernel_launch(void* const* d_in, const int* in_sizes, int n_in,
                              void* d_out, int out_size, void* d_ws, size_t ws_size,
                              hipStream_t stream)
{
    (void)in_sizes; (void)n_in; (void)out_size; (void)ws_size;
    const float* audio = (const float*)d_in[0];
    const float* video = (const float*)d_in[1];
    const float* Wq    = (const float*)d_in[2];
    const float* bq    = (const float*)d_in[3];
    const float* Wk    = (const float*)d_in[4];
    const float* bk    = (const float*)d_in[5];
    const float* Wv    = (const float*)d_in[6];
    const float* bv    = (const float*)d_in[7];
    const float* Wo    = (const float*)d_in[8];
    const float* bo    = (const float*)d_in[9];
    const float* gamma = (const float*)d_in[10];
    const float* beta  = (const float*)d_in[11];

    const long long DD  = (long long)DIM * DIM;       // 1M
    const long long BTD = (long long)NB * TSEQ * DIM; // 32M

    _Float16* Wq16  = (_Float16*)d_ws;
    _Float16* Wk16  = Wq16 + DD;
    _Float16* Wv16  = Wk16 + DD;
    _Float16* Wo16  = Wv16 + DD;
    _Float16* Q16   = Wo16 + DD;          // [B,T,D] row-major
    _Float16* Kt16  = Q16  + BTD;         // [B,D,T] (K transposed)
    _Float16* Vt16  = Kt16 + BTD;         // [B,D,T] (V transposed)
    _Float16* KVt16 = Vt16 + BTD;         // [B,D,D] (KV transposed: [e][d])
    float*    Ksum  = (float*)(KVt16 + (long long)NB * DD);
    _Float16* A16   = (_Float16*)(Ksum + NB * DIM);   // audio f16 [B,T,D]
    _Float16* V16   = A16 + BTD;                      // video f16 [B,T,D]
    _Float16* U16   = Kt16;               // reuse: Kt dead after KV gemm

    const long long sTD = (long long)TSEQ * DIM;   // per-batch [T,D] stride
    const long long sDD = DD;                      // per-batch [D,D] stride

    const size_t sh = 2 * 20480;          // 40 KB double-buffered staging (restage reuses)

    // 1) weights + activations -> f16
    cvt_f16<<<1024, 256, 0, stream>>>(Wq, Wq16, DIM * DIM / 4);
    cvt_f16<<<1024, 256, 0, stream>>>(Wk, Wk16, DIM * DIM / 4);
    cvt_f16<<<1024, 256, 0, stream>>>(Wv, Wv16, DIM * DIM / 4);
    cvt_f16<<<1024, 256, 0, stream>>>(Wo, Wo16, DIM * DIM / 4);
    cvt_f16<<<32768, 256, 0, stream>>>(audio, A16, (int)(BTD / 4));
    cvt_f16<<<32768, 256, 0, stream>>>(video, V16, (int)(BTD / 4));

    dim3 blk(256);
    // 2) projections (per batch: M=T=4096 -> grid.y=32, N=1024 -> grid.x=8)
    gemm_wmma<0><<<dim3(8, 32, NB), blk, sh, stream>>>(
        A16, sTD, DIM, Wq16, 0, DIM, Q16, sTD, DIM, bq, nullptr, 0, DIM / 32);
    gemm_wmma<1><<<dim3(8, 32, NB), blk, sh, stream>>>(
        V16, sTD, DIM, Wk16, 0, DIM, Kt16, sTD, TSEQ, bk, nullptr, 0, DIM / 32);
    gemm_wmma<2><<<dim3(8, 32, NB), blk, sh, stream>>>(
        V16, sTD, DIM, Wv16, 0, DIM, Vt16, sTD, TSEQ, bv, nullptr, 0, DIM / 32);

    // 3) K_sum
    ksum_kernel<<<NB * DIM / 8, 256, 0, stream>>>(Kt16, Ksum);

    // 4) KV = K^T V per batch (M=D, N=D, K=T)
    gemm_wmma<3><<<dim3(8, 8, NB), blk, sh, stream>>>(
        Kt16, sTD, TSEQ, Vt16, sTD, TSEQ, KVt16, sDD, DIM, nullptr, nullptr, 0, TSEQ / 32);

    // 5) attn = (Q @ KV) / (Q*Ksum + eps)   (M=T, N=D, K=D)
    gemm_wmma<4><<<dim3(8, 32, NB), blk, sh, stream>>>(
        Q16, sTD, DIM, KVt16, sDD, DIM, U16, sTD, DIM, nullptr, Ksum, DIM, DIM / 32);

    // 6) Z = U @ Wo^T + bo + audio  -> d_out (f32)
    gemm_wmma<5><<<dim3(8, 32, NB), blk, sh, stream>>>(
        U16, sTD, DIM, Wo16, 0, DIM, d_out, sTD, DIM, bo, audio, sTD, DIM / 32);

    // 7) in-place LayerNorm on d_out
    ln_kernel<<<NB * TSEQ / 8, 256, 0, stream>>>((float*)d_out, gamma, beta);
}
